// MultiScaleExpertCompanion_46523085750351
// MI455X (gfx1250) — matrix operand order repaired and checked by reading, not compiled, not run on hardware
//
#include <hip/hip_runtime.h>
#include <hip/hip_bf16.h>
#include <math.h>

// ---------------------------------------------------------------------------
// MI455X (gfx1250) implementation.
//  - GEMMs: v_wmma_f32_16x16x32_bf16, LDS-staged tiles, wave32, 8 waves/block
//  - A tiles staged in fragment-ready permuted order -> ds_load_b128 frags
//  - B tiles DMA'd by the Tensor Data Mover (tensor_load_to_lds), DOUBLE-
//    BUFFERED: tile k+1 is issued before computing tile k and the wave waits
//    tensorcnt<=1 (TDM completes in-order per wave), so the DMA overlaps the
//    WMMA stream; full drain (tensorcnt<=0) only on the last K-step.
//  - irregular top-k / gathered attention: VALU + LDS
//  - out-projection folded into pooled mean (linearity)
// ---------------------------------------------------------------------------

typedef __bf16 bf16_t;
typedef __attribute__((ext_vector_type(16))) __bf16 v16bf;
typedef __attribute__((ext_vector_type(8)))  float  v8f;

__device__ __forceinline__ float gelu_exact(float x) {
  return 0.5f * x * (1.0f + erff(x * 0.70710678118654752440f));
}

// ---------------------------------------------------------------------------
// WMMA fragment helpers.
// A is staged PERMUTED: As[lane][slot] (32 lanes x 16 bf16), so a fragment is
// two contiguous 16B LDS reads per lane. Mapping from tile coords (m,k):
//   h = (k>>3)&1; lane = h*16+m; v = ((k>>4)<<2)|((k&6)>>1); slot = 2v+(k&1)
// B tile is row-major [32][ldb]: lane gathers column n = lane%16, k in table order.
// ---------------------------------------------------------------------------
__device__ __forceinline__ v16bf frag_a_lds(const bf16_t* Ap) {
  const int lane = threadIdx.x & 31;
  union { uint4 q[2]; v16bf v; } u;
  const uint4* p = (const uint4*)(Ap + lane * 16);
  u.q[0] = p[0];
  u.q[1] = p[1];
  return u.v;
}

__device__ __forceinline__ v16bf frag_b_lds(const bf16_t* B, int ldb) {
  const int lane = threadIdx.x & 31;
  const int n = lane & 15, h = lane >> 4;
  v16bf v;
#pragma unroll
  for (int s = 0; s < 16; ++s) {
    const int k = (s < 8) ? (8 * h + s) : (16 + 8 * h + (s - 8));
    v[s] = B[k * ldb + n];
  }
  return v;
}

__device__ __forceinline__ v8f wmma_bf16(v16bf a, v16bf b, v8f c) {
  return __builtin_amdgcn_wmma_f32_16x16x32_bf16(false, a, false, b,
                                                 (short)0, c, false, false);
}

// ---------------------------------------------------------------------------
// Tensor Data Mover: 2-D bf16 tile (tile_rows x tile_cols) from global to
// LDS; TDM inserts (pad_dwords_m1+1) dwords of padding every 2^(pic+1) dwords
// so the LDS row stride matches the padded tile the fragments read.
// D# layout per CDNA5 ISA; 6-arg builtin.
// ---------------------------------------------------------------------------
#if __has_builtin(__builtin_amdgcn_tensor_load_to_lds)
#define HAVE_TDM 1
typedef unsigned int u32x4 __attribute__((ext_vector_type(4)));
typedef int          i32x8 __attribute__((ext_vector_type(8)));
typedef int          i32x4 __attribute__((ext_vector_type(4)));

__device__ __forceinline__ void tdm_load_tile_bf16(
    unsigned lds_byte_off, const void* gsrc, unsigned tile_cols,
    unsigned tile_rows, unsigned long long row_stride_elems,
    unsigned pad_interval_code, unsigned pad_dwords_m1) {
  const unsigned long long ga = (unsigned long long)gsrc;
  u32x4 g0;
  g0[0] = 1u;                                       // count=1, load, no gather
  g0[1] = lds_byte_off;                             // LDS byte address
  g0[2] = (unsigned)(ga & 0xFFFFFFFFull);           // global_addr[31:0]
  g0[3] = ((unsigned)(ga >> 32) & 0x01FFFFFFu) | 0x80000000u;  // [56:32]|type=2
  i32x8 g1;
  g1[0] = (int)((1u << 16)                          // data_size = 2B
                | (1u << 20)                        // pad_enable
                | (pad_interval_code << 22) | (pad_dwords_m1 << 25));
  g1[1] = (int)((tile_cols & 0xFFFFu) << 16);       // tensor_dim0[15:0]
  g1[2] = (int)((tile_cols >> 16) | ((tile_rows & 0xFFFFu) << 16));
  g1[3] = (int)((tile_rows >> 16) | (tile_cols << 16));  // tile_dim0
  g1[4] = (int)(tile_rows & 0xFFFFu);               // tile_dim1
  g1[5] = (int)(row_stride_elems & 0xFFFFFFFFull);  // dim0_stride[31:0]
  g1[6] = (int)((row_stride_elems >> 32) & 0xFFFFull);
  g1[7] = 0;
  const i32x4 z4 = {0, 0, 0, 0};
  const i32x8 z8 = {0, 0, 0, 0, 0, 0, 0, 0};
  __builtin_amdgcn_tensor_load_to_lds(g0, g1, z4, z4, z8, 0);
}
#else
#define HAVE_TDM 0
#endif

// low 32 bits of a generic LDS-aperture address == LDS byte offset
__device__ __forceinline__ unsigned lds_off(const void* p) {
  return (unsigned)(unsigned long long)p;
}

// ---------------------------------------------------------------------------
// K1: fp32 -> bf16 weight conversion (proj_W 1024x512, qkv_W 512x1536)
// ---------------------------------------------------------------------------
__global__ void __launch_bounds__(256)
k_cvt(const float* __restrict__ wp, const float* __restrict__ wq,
      bf16_t* __restrict__ wp_bf, bf16_t* __restrict__ wq_bf) {
  const size_t i = (size_t)blockIdx.x * 256 + threadIdx.x;
  if (i < (size_t)1024 * 512) wp_bf[i] = (bf16_t)wp[i];
  if (i < (size_t)512 * 1536) wq_bf[i] = (bf16_t)wq[i];
}

// ---------------------------------------------------------------------------
// K2: pentachora centroids -> normalized -> stored transposed as bf16
// ---------------------------------------------------------------------------
__global__ void __launch_bounds__(256)
k_cent(const float* __restrict__ pent, bf16_t* __restrict__ centT) {
  __shared__ float red[256];
  __shared__ float sc_s;
  const int p = blockIdx.x;
  const int tid = threadIdx.x;
  const int d0 = tid, d1 = tid + 256;
  float c0 = 0.f, c1 = 0.f;
#pragma unroll
  for (int v = 0; v < 5; ++v) {
    c0 += pent[((size_t)p * 5 + v) * 512 + d0];
    c1 += pent[((size_t)p * 5 + v) * 512 + d1];
  }
  c0 *= 0.2f; c1 *= 0.2f;
  red[tid] = c0 * c0 + c1 * c1;
  __syncthreads();
  for (int off = 128; off > 0; off >>= 1) {
    if (tid < off) red[tid] += red[tid + off];
    __syncthreads();
  }
  if (tid == 0) sc_s = 1.f / fmaxf(sqrtf(red[0]), 1e-12f);
  __syncthreads();
  const float sc = sc_s;
  centT[(size_t)d0 * 1024 + p] = (bf16_t)(c0 * sc);
  centT[(size_t)d1 * 1024 + p] = (bf16_t)(c1 * sc);
}

// A-staging mapping: thread == pair index (256 pairs of a 16x32 tile)
struct AMap {
  int m, k, slot32;  // slot32: dword index into permuted As
};
__device__ __forceinline__ AMap a_map(int tid) {
  AMap r;
  r.m = tid >> 4;
  r.k = (tid & 15) * 2;
  const int h = (r.k >> 3) & 1;
  const int v = ((r.k >> 4) << 2) | ((r.k & 6) >> 1);
  r.slot32 = (h * 16 + r.m) * 8 + v;
  return r;
}

// ---------------------------------------------------------------------------
// K3: z = L2norm(gelu(LN(seq @ proj_W + proj_b)))  -> z_bf (4096 x 512)
// M=16, N=512, K=1024. 8 waves, each wave owns 16x64 (4 accumulators).
// (single-buffer B: the 33 KB tile union already aliases the LN scratch)
// ---------------------------------------------------------------------------
__global__ void __launch_bounds__(256)
k_proj(const float* __restrict__ seq, const bf16_t* __restrict__ wproj,
       const float* __restrict__ proj_b, const float* __restrict__ ln_g,
       const float* __restrict__ ln_b, bf16_t* __restrict__ z_bf) {
  constexpr int LDB = 520;  // 512 + 8 bf16 pad (TDM: 4 dwords per 256 dwords)
  constexpr int LDC = 516;
  __shared__ bf16_t As[32 * 16];  // permuted A fragments
  __shared__ union {
    bf16_t b[32 * LDB];
    float  c[16 * LDC];
  } sm;
  __shared__ float red[256];
  __shared__ float red2[256];
  __shared__ float mu_s[16], rs_s[16], nrm_s[16];

  const int row0 = blockIdx.x * 16;
  const int tid  = threadIdx.x;
  const int wv   = tid >> 5;
  const AMap am  = a_map(tid);
  unsigned* As32 = (unsigned*)As;

  v8f acc[4];
#pragma unroll
  for (int t = 0; t < 4; ++t) acc[t] = {0.f, 0.f, 0.f, 0.f, 0.f, 0.f, 0.f, 0.f};

  for (int kk = 0; kk < 1024; kk += 32) {
    // stage A: one float2 load + packed bf16 dword store per thread
    {
      const float2 f =
          *(const float2*)(seq + (size_t)(row0 + am.m) * 1024 + kk + am.k);
      union { bf16_t h[2]; unsigned u; } pk;
      pk.h[0] = (bf16_t)f.x;
      pk.h[1] = (bf16_t)f.y;
      As32[am.slot32] = pk.u;
    }
    // stage B: 32x512 bf16 tile
#if HAVE_TDM
    if (tid < 32) {
      tdm_load_tile_bf16(lds_off(sm.b), wproj + (size_t)kk * 512,
                         /*cols*/512, /*rows*/32, /*stride*/512,
                         /*pad_interval 256dw*/7, /*pad 4dw*/3);
      __builtin_amdgcn_s_wait_tensorcnt(0);
    }
#else
    {
      const int r  = tid >> 3;
      const int c0 = (tid & 7) * 64;
      const uint4* src = (const uint4*)(wproj + (size_t)(kk + r) * 512 + c0);
      uint4* dst = (uint4*)(sm.b + r * LDB + c0);
#pragma unroll
      for (int j = 0; j < 8; ++j) dst[j] = src[j];
    }
#endif
    if (kk + 32 < 1024)
      __builtin_prefetch(wproj + (size_t)(kk + 32) * 512, 0, 1);
    __syncthreads();
    const v16bf a = frag_a_lds(As);
#pragma unroll
    for (int t = 0; t < 4; ++t) {
      const v16bf b = frag_b_lds(sm.b + (wv * 64 + t * 16), LDB);
      acc[t] = wmma_bf16(a, b, acc[t]);
    }
    __syncthreads();
  }

  // dump accumulators: C[r+8h][lane&15]
  {
    const int lane = tid & 31;
    const int n = lane & 15, h = lane >> 4;
#pragma unroll
    for (int t = 0; t < 4; ++t)
#pragma unroll
      for (int r = 0; r < 8; ++r)
        sm.c[(r + 8 * h) * LDC + wv * 64 + t * 16 + n] = acc[t][r];
  }
  __syncthreads();

  // LN over 512 cols per row, 16 threads per row
  const int r = tid >> 4;
  const int j = tid & 15;
  float s1 = 0.f, s2 = 0.f;
  for (int c = j * 32; c < j * 32 + 32; ++c) {
    const float x = sm.c[r * LDC + c] + proj_b[c];
    sm.c[r * LDC + c] = x;
    s1 += x; s2 += x * x;
  }
  red[tid] = s1; red2[tid] = s2;
  __syncthreads();
  for (int off = 8; off > 0; off >>= 1) {
    if (j < off) { red[tid] += red[tid + off]; red2[tid] += red2[tid + off]; }
    __syncthreads();
  }
  if (j == 0) {
    const float mu = red[r * 16] * (1.f / 512.f);
    const float var = red2[r * 16] * (1.f / 512.f) - mu * mu;
    mu_s[r] = mu;
    rs_s[r] = rsqrtf(var + 1e-5f);
  }
  __syncthreads();
  const float mu = mu_s[r], rs = rs_s[r];
  float q = 0.f;
  for (int c = j * 32; c < j * 32 + 32; ++c) {
    const float x = sm.c[r * LDC + c];
    const float g = gelu_exact((x - mu) * rs * ln_g[c] + ln_b[c]);
    sm.c[r * LDC + c] = g;
    q += g * g;
  }
  red[tid] = q;
  __syncthreads();
  for (int off = 8; off > 0; off >>= 1) {
    if (j < off) red[tid] += red[tid + off];
    __syncthreads();
  }
  if (j == 0) nrm_s[r] = 1.f / fmaxf(sqrtf(red[r * 16]), 1e-12f);
  __syncthreads();
  const float sc = nrm_s[r];
  for (int c = j * 32; c < j * 32 + 32; ++c)
    z_bf[(size_t)(row0 + r) * 512 + c] = (bf16_t)(sm.c[r * LDC + c] * sc);
}

// ---------------------------------------------------------------------------
// K4: anchor scores z @ cent^T (4096 x 1024, K=512), fused row argmax,
//     gather shared_positions -> pos_out (4096).
//     Double-buffered TDM pipeline over the 16 K-steps of each N-chunk.
// ---------------------------------------------------------------------------
__global__ void __launch_bounds__(256)
k_anchor(const bf16_t* __restrict__ z_bf, const bf16_t* __restrict__ centT,
         const float* __restrict__ positions, float* __restrict__ pos_out) {
  constexpr int LDB = 136, LDS_ = 132;
  __shared__ bf16_t As[32 * 16];
  __shared__ bf16_t Bs[2][32 * LDB];
  __shared__ float  Ss[16 * LDS_];
  __shared__ float  redv[256];
  __shared__ int    redi[256];

  const int row0 = blockIdx.x * 16;
  const int tid  = threadIdx.x;
  const int wv   = tid >> 5;
  const AMap am  = a_map(tid);
  unsigned* As32 = (unsigned*)As;
  const unsigned* zsrc = (const unsigned*)z_bf;

  float bestv = -3.4e38f;
  int   besti = 0;

  for (int p0 = 0; p0 < 1024; p0 += 128) {
    v8f acc = {0.f, 0.f, 0.f, 0.f, 0.f, 0.f, 0.f, 0.f};
#if HAVE_TDM
    if (tid < 32)  // prologue: tile kk=0 into buffer 0
      tdm_load_tile_bf16(lds_off(Bs[0]), centT + p0,
                         128, 32, 1024, /*pic 64dw*/5, /*pad 4dw*/3);
#endif
    for (int kk = 0; kk < 512; kk += 32) {
      const int idx = (kk >> 5) & 1;
      As32[am.slot32] = zsrc[((size_t)(row0 + am.m) * 512 + kk + am.k) >> 1];
#if HAVE_TDM
      if (tid < 32) {
        if (kk + 32 < 512) {  // issue next tile, wait only for the older one
          tdm_load_tile_bf16(lds_off(Bs[idx ^ 1]),
                             centT + (size_t)(kk + 32) * 1024 + p0,
                             128, 32, 1024, 5, 3);
          __builtin_amdgcn_s_wait_tensorcnt(1);
        } else {
          __builtin_amdgcn_s_wait_tensorcnt(0);
        }
      }
#else
      {
        const int r  = tid >> 3;
        const int c0 = (tid & 7) * 16;
        const uint4* src =
            (const uint4*)(centT + (size_t)(kk + r) * 1024 + p0 + c0);
        uint4* dst = (uint4*)(Bs[idx] + r * LDB + c0);
        dst[0] = src[0];
        dst[1] = src[1];
      }
#endif
      __syncthreads();
      const v16bf a = frag_a_lds(As);
      const v16bf b = frag_b_lds(Bs[idx] + wv * 16, LDB);
      acc = wmma_bf16(a, b, acc);
      __syncthreads();
    }
    {
      const int lane = tid & 31;
      const int n = lane & 15, h = lane >> 4;
#pragma unroll
      for (int r = 0; r < 8; ++r)
        Ss[(r + 8 * h) * LDS_ + wv * 16 + n] = acc[r];
    }
    __syncthreads();
    {
      const int r = tid >> 4, jj = tid & 15;
      for (int c = jj * 8; c < jj * 8 + 8; ++c) {
        const float v = Ss[r * LDS_ + c];
        if (v > bestv) { bestv = v; besti = p0 + c; }  // strict > keeps first
      }
    }
    __syncthreads();
  }

  redv[tid] = bestv; redi[tid] = besti;
  __syncthreads();
  const int r = tid >> 4, jj = tid & 15;
  for (int off = 8; off > 0; off >>= 1) {
    if (jj < off) {
      const float v2 = redv[tid + off];
      const int   i2 = redi[tid + off];
      if (v2 > redv[tid] || (v2 == redv[tid] && i2 < redi[tid])) {
        redv[tid] = v2; redi[tid] = i2;
      }
    }
    __syncthreads();
  }
  if (jj == 0) pos_out[row0 + r] = positions[redi[tid]];
}

// ---------------------------------------------------------------------------
// K5: per-query top-128 nearest |pos_q - pos_s| (ties -> lowest index)
// ---------------------------------------------------------------------------
__global__ void __launch_bounds__(256)
k_topk(const float* __restrict__ pos, int* __restrict__ routes) {
  __shared__ float dist[2048];
  __shared__ float redv[256];
  __shared__ int   redi[256];
  const int q = blockIdx.x;
  const int b = blockIdx.y;
  const int tid = threadIdx.x;
  const float pq = pos[b * 2048 + q];
  for (int s = tid; s < 2048; s += 256) dist[s] = fabsf(pq - pos[b * 2048 + s]);
  __syncthreads();
  int* rt = routes + (size_t)(b * 2048 + q) * 128;
  for (int it = 0; it < 128; ++it) {
    float bv = 3.4e38f; int bi = 0x7fffffff;
    for (int s = tid; s < 2048; s += 256) {
      const float v = dist[s];
      if (v < bv || (v == bv && s < bi)) { bv = v; bi = s; }
    }
    redv[tid] = bv; redi[tid] = bi;
    __syncthreads();
    for (int off = 128; off > 0; off >>= 1) {
      if (tid < off) {
        const float v2 = redv[tid + off];
        const int   i2 = redi[tid + off];
        if (v2 < redv[tid] || (v2 == redv[tid] && i2 < redi[tid])) {
          redv[tid] = v2; redi[tid] = i2;
        }
      }
      __syncthreads();
    }
    if (tid == 0) { rt[it] = redi[0]; dist[redi[0]] = 3.4e38f; }
    __syncthreads();
  }
}

// ---------------------------------------------------------------------------
// K6: qkv = z @ qkv_W + qkv_b  (4096 x 1536, K=512), fp32 out.
//     Double-buffered TDM pipeline over the 16 K-steps.
// ---------------------------------------------------------------------------
__global__ void __launch_bounds__(256)
k_qkv(const bf16_t* __restrict__ z_bf, const bf16_t* __restrict__ wq,
      const float* __restrict__ qkv_b, float* __restrict__ qkv) {
  constexpr int LDB = 136;
  __shared__ bf16_t As[32 * 16];
  __shared__ bf16_t Bs[2][32 * LDB];
  const int row0 = blockIdx.x * 16;
  const int n0g  = blockIdx.y * 128;
  const int tid  = threadIdx.x;
  const int wv   = tid >> 5;
  const AMap am  = a_map(tid);
  unsigned* As32 = (unsigned*)As;
  const unsigned* zsrc = (const unsigned*)z_bf;

  v8f acc = {0.f, 0.f, 0.f, 0.f, 0.f, 0.f, 0.f, 0.f};
#if HAVE_TDM
  if (tid < 32)  // prologue: tile kk=0 into buffer 0
    tdm_load_tile_bf16(lds_off(Bs[0]), wq + n0g, 128, 32, 1536, 5, 3);
#endif
  for (int kk = 0; kk < 512; kk += 32) {
    const int idx = (kk >> 5) & 1;
    As32[am.slot32] = zsrc[((size_t)(row0 + am.m) * 512 + kk + am.k) >> 1];
#if HAVE_TDM
    if (tid < 32) {
      if (kk + 32 < 512) {  // overlap: issue next tile, wait for older only
        tdm_load_tile_bf16(lds_off(Bs[idx ^ 1]),
                           wq + (size_t)(kk + 32) * 1536 + n0g,
                           128, 32, 1536, 5, 3);
        __builtin_amdgcn_s_wait_tensorcnt(1);
      } else {
        __builtin_amdgcn_s_wait_tensorcnt(0);
      }
    }
#else
    {
      const int r  = tid >> 3;
      const int c0 = (tid & 7) * 16;
      const uint4* src = (const uint4*)(wq + (size_t)(kk + r) * 1536 + n0g + c0);
      uint4* dst = (uint4*)(Bs[idx] + r * LDB + c0);
      dst[0] = src[0];
      dst[1] = src[1];
    }
#endif
    __syncthreads();
    const v16bf a = frag_a_lds(As);
    const v16bf b = frag_b_lds(Bs[idx] + wv * 16, LDB);
    acc = wmma_bf16(a, b, acc);
    __syncthreads();
  }
  const int lane = tid & 31;
  const int n = lane & 15, h = lane >> 4;
  const int col = n0g + wv * 16 + n;
  const float bias = qkv_b[col];
#pragma unroll
  for (int r = 0; r < 8; ++r)
    qkv[(size_t)(row0 + r + 8 * h) * 1536 + col] = acc[r] + bias;
}

// ---------------------------------------------------------------------------
// K7: gathered neighborhood attention, one block per (b, h, q).
// ---------------------------------------------------------------------------
__global__ void __launch_bounds__(128)
k_attn(const float* __restrict__ qkv, const int* __restrict__ routes,
       float* __restrict__ attnout) {
  __shared__ float qv[64];
  __shared__ float sc[128];
  __shared__ int   rt[128];
  __shared__ float red[128];
  const int q = blockIdx.x, h = blockIdx.y, b = blockIdx.z;
  const int row = b * 2048 + q;
  const int tid = threadIdx.x;

  rt[tid] = routes[(size_t)row * 128 + tid];
  if (tid < 64) qv[tid] = qkv[(size_t)row * 1536 + h * 64 + tid];
  __syncthreads();

  {
    const float* kp = qkv + (size_t)(b * 2048 + rt[tid]) * 1536 + 512 + h * 64;
    float d = 0.f;
#pragma unroll
    for (int i = 0; i < 64; ++i) d += qv[i] * kp[i];
    sc[tid] = d * 0.125f;  // 1/sqrt(64)
  }
  red[tid] = sc[tid];
  __syncthreads();
  for (int off = 64; off > 0; off >>= 1) {
    if (tid < off) red[tid] = fmaxf(red[tid], red[tid + off]);
    __syncthreads();
  }
  const float mx = red[0];
  __syncthreads();
  const float e = expf(sc[tid] - mx);
  sc[tid] = e;
  red[tid] = e;
  __syncthreads();
  for (int off = 64; off > 0; off >>= 1) {
    if (tid < off) red[tid] += red[tid + off];
    __syncthreads();
  }
  const float inv = 1.f / red[0];
  __syncthreads();

  const int d = tid & 63, half = tid >> 6;
  float accv = 0.f;
  for (int k2 = half * 64; k2 < half * 64 + 64; ++k2)
    accv += sc[k2] * qkv[(size_t)(b * 2048 + rt[k2]) * 1536 + 1024 + h * 64 + d];
  red[tid] = accv;
  __syncthreads();
  if (tid < 64)
    attnout[(size_t)row * 512 + h * 64 + d] = (red[tid] + red[tid + 64]) * inv;
}

// ---------------------------------------------------------------------------
// K8: pooledA[b][d] = mean_s attnout[b][s][d]
// ---------------------------------------------------------------------------
__global__ void __launch_bounds__(512)
k_pool(const float* __restrict__ attnout, float* __restrict__ pooledA) {
  const int b = blockIdx.x;
  const int d = threadIdx.x;
  float s = 0.f;
  for (int t = 0; t < 2048; ++t)
    s += attnout[(size_t)(b * 2048 + t) * 512 + d];
  pooledA[b * 512 + d] = s * (1.f / 2048.f);
}

// ---------------------------------------------------------------------------
// K9: pooled = pooledA @ out_W + out_b, then three LN+GELU MLP heads
// ---------------------------------------------------------------------------
__device__ __forceinline__ float block_reduce_sum512(float v, float* red) {
  const int tid = threadIdx.x;
  red[tid] = v;
  __syncthreads();
  for (int off = 256; off > 0; off >>= 1) {
    if (tid < off) red[tid] += red[tid + off];
    __syncthreads();
  }
  const float r = red[0];
  __syncthreads();
  return r;
}

__device__ void head_scale(const float* pooled, float* hbuf, float* red,
                           int hid, int sdim, int ooff,
                           const float* w1, const float* b1,
                           const float* g, const float* bb,
                           const float* w2, const float* b2,
                           float* out) {
  const int tid = threadIdx.x;
  for (int b = 0; b < 2; ++b)
    for (int c = tid; c < hid; c += 512) {
      float s = b1[c];
      for (int k = 0; k < 512; ++k)
        s += pooled[b * 512 + k] * w1[(size_t)k * hid + c];
      hbuf[b * 1024 + c] = s;
    }
  __syncthreads();
  for (int b = 0; b < 2; ++b) {
    float s1 = 0.f, s2 = 0.f;
    for (int c = tid; c < hid; c += 512) {
      const float x = hbuf[b * 1024 + c];
      s1 += x; s2 += x * x;
    }
    const float tot1 = block_reduce_sum512(s1, red);
    const float tot2 = block_reduce_sum512(s2, red);
    const float mu = tot1 / (float)hid;
    const float var = tot2 / (float)hid - mu * mu;
    const float rs = rsqrtf(var + 1e-5f);
    for (int c = tid; c < hid; c += 512) {
      const float x = hbuf[b * 1024 + c];
      hbuf[b * 1024 + c] = gelu_exact((x - mu) * rs * g[c] + bb[c]);
    }
    __syncthreads();
  }
  for (int b = 0; b < 2; ++b)
    for (int j = tid; j < sdim; j += 512) {
      float s = b2[j];
      for (int k = 0; k < hid; ++k)
        s += hbuf[b * 1024 + k] * w2[(size_t)k * sdim + j];
      out[b * 896 + ooff + j] = s;
    }
  __syncthreads();
}

__global__ void __launch_bounds__(512)
k_heads(const float* __restrict__ pooledA,
        const float* __restrict__ outW, const float* __restrict__ outb,
        const float* w1_128, const float* b1_128, const float* g_128,
        const float* bb_128, const float* w2_128, const float* b2_128,
        const float* w1_256, const float* b1_256, const float* g_256,
        const float* bb_256, const float* w2_256, const float* b2_256,
        const float* w1_512, const float* b1_512, const float* g_512,
        const float* bb_512, const float* w2_512, const float* b2_512,
        float* __restrict__ out) {
  __shared__ float pooled[1024];
  __shared__ float hbuf[2048];
  __shared__ float red[512];
  const int tid = threadIdx.x;
  for (int b = 0; b < 2; ++b) {
    float s = outb[tid];
    for (int k = 0; k < 512; ++k)
      s += pooledA[b * 512 + k] * outW[(size_t)k * 512 + tid];
    pooled[b * 512 + tid] = s;
  }
  __syncthreads();
  head_scale(pooled, hbuf, red, 256, 128, 0,
             w1_128, b1_128, g_128, bb_128, w2_128, b2_128, out);
  head_scale(pooled, hbuf, red, 512, 256, 128,
             w1_256, b1_256, g_256, bb_256, w2_256, b2_256, out);
  head_scale(pooled, hbuf, red, 1024, 512, 384,
             w1_512, b1_512, g_512, bb_512, w2_512, b2_512, out);
}

// ---------------------------------------------------------------------------
// launch
// ---------------------------------------------------------------------------
extern "C" void kernel_launch(void* const* d_in, const int* in_sizes, int n_in,
                              void* d_out, int out_size, void* d_ws,
                              size_t ws_size, hipStream_t stream) {
  (void)in_sizes; (void)n_in; (void)out_size; (void)ws_size;
  const float* seq   = (const float*)d_in[0];
  const float* pent  = (const float*)d_in[1];
  const float* spos  = (const float*)d_in[2];
  const float* projW = (const float*)d_in[3];
  const float* projb = (const float*)d_in[4];
  const float* lng   = (const float*)d_in[5];
  const float* lnb   = (const float*)d_in[6];
  const float* qkvW  = (const float*)d_in[7];
  const float* qkvb  = (const float*)d_in[8];
  const float* outW  = (const float*)d_in[9];
  const float* outb  = (const float*)d_in[10];
  const float* w1a = (const float*)d_in[11]; const float* b1a = (const float*)d_in[12];
  const float* ga  = (const float*)d_in[13]; const float* bba = (const float*)d_in[14];
  const float* w2a = (const float*)d_in[15]; const float* b2a = (const float*)d_in[16];
  const float* w1b = (const float*)d_in[17]; const float* b1b = (const float*)d_in[18];
  const float* gb  = (const float*)d_in[19]; const float* bbb = (const float*)d_in[20];
  const float* w2b = (const float*)d_in[21]; const float* b2b = (const float*)d_in[22];
  const float* w1c = (const float*)d_in[23]; const float* b1c = (const float*)d_in[24];
  const float* gc  = (const float*)d_in[25]; const float* bbc = (const float*)d_in[26];
  const float* w2c = (const float*)d_in[27]; const float* b2c = (const float*)d_in[28];
  float* out = (float*)d_out;

  char* ws = (char*)d_ws;
  bf16_t* wproj_bf = (bf16_t*)(ws + 0);          // 1,048,576 B
  bf16_t* wqkv_bf  = (bf16_t*)(ws + 1048576);    // 1,572,864 B
  bf16_t* centT    = (bf16_t*)(ws + 2621440);    // 1,048,576 B
  bf16_t* z_bf     = (bf16_t*)(ws + 3670016);    // 4,194,304 B
  float*  posb     = (float*) (ws + 7864320);    //    16,384 B
  int*    routes   = (int*)   (ws + 7880704);    // 2,097,152 B
  float*  qkvbuf   = (float*) (ws + 9977856);    // 25,165,824 B
  float*  attnout  = (float*) (ws + 35143680);   // 8,388,608 B
  float*  pooledA  = (float*) (ws + 43532288);   //     4,096 B

  k_cvt   <<<3072, 256, 0, stream>>>(projW, qkvW, wproj_bf, wqkv_bf);
  k_cent  <<<1024, 256, 0, stream>>>(pent, centT);
  k_proj  <<<256, 256, 0, stream>>>(seq, wproj_bf, projb, lng, lnb, z_bf);
  k_anchor<<<256, 256, 0, stream>>>(z_bf, centT, spos, posb);
  k_topk  <<<dim3(2048, 2), 256, 0, stream>>>(posb, routes);
  k_qkv   <<<dim3(256, 12), 256, 0, stream>>>(z_bf, wqkv_bf, qkvb, qkvbuf);
  k_attn  <<<dim3(2048, 8, 2), 128, 0, stream>>>(qkvbuf, routes, attnout);
  k_pool  <<<2, 512, 0, stream>>>(attnout, pooledA);
  k_heads <<<1, 512, 0, stream>>>(pooledA, outW, outb,
                                  w1a, b1a, ga, bba, w2a, b2a,
                                  w1b, b1b, gb, bbb, w2b, b2b,
                                  w1c, b1c, gc, bbc, w2c, b2c, out);
}
